// LOL_75127567942100
// MI455X (gfx1250) — matrix operand order repaired and checked by reading, not compiled, not run on hardware
//
#include <hip/hip_runtime.h>
#include <hip/hip_bf16.h>

// MI455X / gfx1250 wave32. Memory-bound (~450MB @ 23.3TB/s => ~19us floor).
// GEMMs: v_wmma_f32_16x16x32_f16, f32 accumulate. All fragment loads are
// unconditional aligned global_load_b128; each wave runs MT=2 M-tiles per
// N-tile so B fragments are register-reused and WMMA hazards overlap.

typedef __attribute__((ext_vector_type(8)))  _Float16 v8h;
typedef __attribute__((ext_vector_type(16))) _Float16 v16h;
typedef __attribute__((ext_vector_type(8)))  float    v8f;

#define EPSBN 1e-5f

// ---------------------------------------------------------------------------
// Per-row attention (collapsed): out[q] = c2 * sum_k softmax_q(c1*x_q*x_k)[q]*x_k
// c1 = (Wq.Wk)/4, c2 = (Wv.Wo). Softmax over axis=1 (q) per column k.
// Input: f32, row stride inStride (only cols 0..9 read).
// Output: f16, row stride 32, cols 10..31 zero-filled (feeds K=32 WMMA GEMM).
// ---------------------------------------------------------------------------
__global__ void attn_rows_kernel(const float* __restrict__ in, int inStride,
                                 const float* __restrict__ Wq, const float* __restrict__ Wk,
                                 const float* __restrict__ Wv, const float* __restrict__ Wo,
                                 _Float16* __restrict__ out, int B) {
  int row = blockIdx.x * blockDim.x + threadIdx.x;
  if (row >= B) return;
  float c1 = 0.f, c2 = 0.f;
#pragma unroll
  for (int i = 0; i < 16; ++i) { c1 += Wq[i] * Wk[i]; c2 += Wv[i] * Wo[i]; }
  c1 *= 0.25f;

  float xv[10];
#pragma unroll
  for (int q = 0; q < 10; ++q) xv[q] = in[(size_t)row * inStride + q];
  float o[10];
#pragma unroll
  for (int q = 0; q < 10; ++q) o[q] = 0.f;

#pragma unroll
  for (int k = 0; k < 10; ++k) {
    float a = c1 * xv[k];
    float m = -3.402823466e38f;
#pragma unroll
    for (int q = 0; q < 10; ++q) m = fmaxf(m, a * xv[q]);
    float e[10]; float s = 0.f;
#pragma unroll
    for (int q = 0; q < 10; ++q) { e[q] = __expf(a * xv[q] - m); s += e[q]; }
    float f = xv[k] / s;
#pragma unroll
    for (int q = 0; q < 10; ++q) o[q] += e[q] * f;
  }

  v8h v0 = {}, v1 = {}, v2 = {}, v3 = {};
#pragma unroll
  for (int q = 0; q < 8; ++q)  v0[q] = (_Float16)(c2 * o[q]);
  v1[0] = (_Float16)(c2 * o[8]);
  v1[1] = (_Float16)(c2 * o[9]);
  _Float16* orow = out + (size_t)row * 32;   // 64B-aligned
  *(v8h*)(orow +  0) = v0;
  *(v8h*)(orow +  8) = v1;
  *(v8h*)(orow + 16) = v2;
  *(v8h*)(orow + 24) = v3;
}

// ---------------------------------------------------------------------------
// Weight prep: W (K x N, f32) -> WT (Npad x Kpad, f16, zero-padded), bias pad.
// ---------------------------------------------------------------------------
__global__ void prep_w_kernel(const float* __restrict__ W, const float* __restrict__ bias,
                              _Float16* __restrict__ WT, float* __restrict__ biasPad,
                              int K, int N, int Kpad, int Npad) {
  int idx = blockIdx.x * blockDim.x + threadIdx.x;
  if (idx < Kpad * Npad) {
    int n = idx / Kpad, k = idx % Kpad;
    float v = (k < K && n < N) ? W[(size_t)k * N + n] : 0.f;
    WT[idx] = (_Float16)v;
  }
  if (idx < Npad) biasPad[idx] = (idx < N) ? bias[idx] : 0.f;
}

// ---------------------------------------------------------------------------
// (sum,sumsq) -> f16 BN coefficients: y = a*z + c
// ---------------------------------------------------------------------------
__global__ void bn_coef_kernel(const float* __restrict__ gsum, const float* __restrict__ gssq,
                               const float* __restrict__ gamma, const float* __restrict__ beta,
                               _Float16* __restrict__ ca, _Float16* __restrict__ cc,
                               int N, float invB) {
  int i = blockIdx.x * blockDim.x + threadIdx.x;
  if (i < N) {
    float mu  = gsum[i] * invB;
    float var = gssq[i] * invB - mu * mu;   // biased variance
    float a   = gamma[i] * rsqrtf(var + EPSBN);
    ca[i] = (_Float16)a;
    cc[i] = (_Float16)(beta[i] - a * mu);
  }
}

// ---------------------------------------------------------------------------
// Skinny GEMM: Z = act(A) @ W + bias. MT 16x16 tiles per wave (shared B frag);
// 8 waves/block share one N-tile (LDS stats reduction).
//   A : (B x LDA) f16, rows 16B-aligned; act = identity or relu(ca*z+cc)
//   WT: (Npad x K) f16, lane reads k runs [kc+8hi,+8) and [kc+16+8hi,+8) -> 2x b128
//   Z : (B x LDZ) TOUT; NSTORE=0 -> store all 16 cols, else mask colB<NSTORE
// K in {32,128,256}, multiple of 32 -> zero masking in the hot loop.
// ---------------------------------------------------------------------------
template<int K, int LDA, int LDZ, int NSTORE, int MT, bool IN_BN, bool OUT_STATS, typename TOUT>
__global__ void __launch_bounds__(256)
gemm_kernel(const _Float16* __restrict__ A,
            const _Float16* __restrict__ ca, const _Float16* __restrict__ cc,
            const _Float16* __restrict__ WT, const float* __restrict__ biasPad,
            TOUT* __restrict__ Z, float* __restrict__ gsum, float* __restrict__ gssq) {
  __shared__ float ssum[16];
  __shared__ float ssqs[16];
  if constexpr (OUT_STATS) {
    if (threadIdx.x < 16) { ssum[threadIdx.x] = 0.f; ssqs[threadIdx.x] = 0.f; }
    __syncthreads();
  }
  const int lane = threadIdx.x & 31;
  const int wave = threadIdx.x >> 5;
  const int hi   = lane >> 4;
  const int l15  = lane & 15;
  const long m0  = ((long)blockIdx.x * 8 + wave) * (16 * MT);
  const int  n0  = blockIdx.y * 16;
  const int  colB = n0 + l15;

  const _Float16* Arow[MT];
#pragma unroll
  for (int mt = 0; mt < MT; ++mt) Arow[mt] = A + (m0 + mt * 16 + l15) * (long)LDA;
  const _Float16* Wrow = WT + (size_t)colB * K;

  constexpr int KC = K / 32;
  const _Float16 hzero = (_Float16)0.0f;

  v8f acc[MT];
#pragma unroll
  for (int mt = 0; mt < MT; ++mt) acc[mt] = (v8f){};

#pragma unroll
  for (int kc0 = 0; kc0 < KC; ++kc0) {
    const int kc = kc0 * 32;
    const int kA = kc + hi * 8;     // A runs: [kA, +8), [kA+16, +8)
    const int kB = kc + hi * 8;     // B runs: [kB, +8), [kB+16, +8)

    // B fragment: loaded once, reused for all MT tiles
    v8h b0 = *(const v8h*)(Wrow + kB);
    v8h b1 = *(const v8h*)(Wrow + kB + 16);
    v16h bf = __builtin_shufflevector(b0, b1, 0,1,2,3,4,5,6,7,8,9,10,11,12,13,14,15);

    v8h ca0, cc0, ca1, cc1;
    if constexpr (IN_BN) {
      ca0 = *(const v8h*)(ca + kA);
      cc0 = *(const v8h*)(cc + kA);
      ca1 = *(const v8h*)(ca + kA + 16);
      cc1 = *(const v8h*)(cc + kA + 16);
    }

#pragma unroll
    for (int mt = 0; mt < MT; ++mt) {
      if (kc0 + 1 < KC)                     // gfx1250 global_prefetch_b8
        __builtin_prefetch((const void*)(Arow[mt] + kc + 32), 0, 3);
      v8h a0 = *(const v8h*)(Arow[mt] + kA);
      v8h a1 = *(const v8h*)(Arow[mt] + kA + 16);
      if constexpr (IN_BN) {
#pragma unroll
        for (int j = 0; j < 8; ++j) {
          _Float16 t0 = ca0[j] * a0[j] + cc0[j];
          _Float16 t1 = ca1[j] * a1[j] + cc1[j];
          a0[j] = t0 > hzero ? t0 : hzero;
          a1[j] = t1 > hzero ? t1 : hzero;
        }
      }
      v16h af = __builtin_shufflevector(a0, a1, 0,1,2,3,4,5,6,7,8,9,10,11,12,13,14,15);
      acc[mt] = __builtin_amdgcn_wmma_f32_16x16x32_f16(
          false, af, false, bf, (short)0, acc[mt], false, false);
    }
  }

  // C/D layout: VGPR r -> (M = tile_m0 + r + 8*hi, N = n0 + l15).
  // One base pointer per tile; row offsets are compile-time -> immediate offsets.
  const float bv = biasPad[colB];
  float lsum = 0.f, lssq = 0.f;
#pragma unroll
  for (int mt = 0; mt < MT; ++mt) {
    TOUT* zp = Z + (m0 + mt * 16 + hi * 8) * (long)LDZ + colB;
#pragma unroll
    for (int r = 0; r < 8; ++r) {
      const float v = acc[mt][r] + bv;
      if constexpr (NSTORE != 0) {
        if (colB < NSTORE) zp[r * LDZ] = (TOUT)v;
      } else {
        zp[r * LDZ] = (TOUT)v;
      }
      lsum += v; lssq += v * v;
    }
  }
  if constexpr (OUT_STATS) {
    atomicAdd(&ssum[l15], lsum);            // ds_add_f32
    atomicAdd(&ssqs[l15], lssq);
    __syncthreads();
    if (threadIdx.x < 16) {
      atomicAdd(&gsum[n0 + threadIdx.x], ssum[threadIdx.x]);
      atomicAdd(&gssq[n0 + threadIdx.x], ssqs[threadIdx.x]);
    }
  }
}

// ---------------------------------------------------------------------------
extern "C" void kernel_launch(void* const* d_in, const int* in_sizes, int n_in,
                              void* d_out, int out_size, void* d_ws, size_t ws_size,
                              hipStream_t stream) {
  const float* x   = (const float*)d_in[0];
  const float* Wq1 = (const float*)d_in[1];
  const float* Wk1 = (const float*)d_in[2];
  const float* Wv1 = (const float*)d_in[3];
  const float* Wo1 = (const float*)d_in[4];
  const float* Wq2 = (const float*)d_in[5];
  const float* Wk2 = (const float*)d_in[6];
  const float* Wv2 = (const float*)d_in[7];
  const float* Wo2 = (const float*)d_in[8];
  const float* W1a = (const float*)d_in[9];
  const float* b1a = (const float*)d_in[10];
  const float* g1a = (const float*)d_in[11];
  const float* be1a= (const float*)d_in[12];
  const float* W1b = (const float*)d_in[13];
  const float* b1b = (const float*)d_in[14];
  const float* g1b = (const float*)d_in[15];
  const float* be1b= (const float*)d_in[16];
  const float* W1c = (const float*)d_in[17];
  const float* b1c = (const float*)d_in[18];
  const float* W2a = (const float*)d_in[19];
  const float* b2a = (const float*)d_in[20];
  const float* g2a = (const float*)d_in[21];
  const float* be2a= (const float*)d_in[22];
  const float* W2b = (const float*)d_in[23];
  const float* b2b = (const float*)d_in[24];
  const float* g2b = (const float*)d_in[25];
  const float* be2b= (const float*)d_in[26];
  const float* W2c = (const float*)d_in[27];
  const float* b2c = (const float*)d_in[28];

  const int B = in_sizes[0] / 10;          // 131072
  const float invB = 1.0f / (float)B;

  // ---- workspace layout ----
  _Float16* hA  = (_Float16*)d_ws;                 // B x 32 f16 (attn out, padded)
  float*    hz  = (float*)(hA + (size_t)B * 32);   // B x 32 f32 (pre-attn2 z)
  _Float16* za  = (_Float16*)(hz + (size_t)B * 32);// B x 128 f16 pre-BN z
  _Float16* zb  = za + (size_t)B * 128;            // B x 256 f16 pre-BN z
  _Float16* wtA1= zb + (size_t)B * 256;            // 128 x 32
  _Float16* wtB1= wtA1 + 128 * 32;                 // 256 x 128
  _Float16* wtC1= wtB1 + 256 * 128;                // 16 x 256
  _Float16* wtA2= wtC1 + 16 * 256;
  _Float16* wtB2= wtA2 + 128 * 32;
  _Float16* wtC2= wtB2 + 256 * 128;
  _Float16* ca  = wtC2 + 16 * 256;                 // 256 f16
  _Float16* cc  = ca + 256;                        // 256 f16
  float* bpA1 = (float*)(cc + 256);                // padded biases (f32)
  float* bpB1 = bpA1 + 128;
  float* bpC1 = bpB1 + 256;
  float* bpA2 = bpC1 + 16;
  float* bpB2 = bpA2 + 128;
  float* bpC2 = bpB2 + 256;
  float* gsm  = bpC2 + 16;                         // 256 sums
  float* gsq  = gsm + 256;                         // 256 sumsqs

  dim3 blk(256);
  dim3 gAttn((B + 255) / 256);
  const int gx = B / 256;                  // 8 waves x (MT=2 x 16 rows) per block

  // ---- weight prep (tiny) ----
  prep_w_kernel<<<16,  blk, 0, stream>>>(W1a, b1a, wtA1, bpA1, 10, 128, 32, 128);
  prep_w_kernel<<<128, blk, 0, stream>>>(W1b, b1b, wtB1, bpB1, 128, 256, 128, 256);
  prep_w_kernel<<<16,  blk, 0, stream>>>(W1c, b1c, wtC1, bpC1, 256, 10, 256, 16);
  prep_w_kernel<<<16,  blk, 0, stream>>>(W2a, b2a, wtA2, bpA2, 10, 128, 32, 128);
  prep_w_kernel<<<128, blk, 0, stream>>>(W2b, b2b, wtB2, bpB2, 128, 256, 128, 256);
  prep_w_kernel<<<16,  blk, 0, stream>>>(W2c, b2c, wtC2, bpC2, 256, 2, 256, 16);

  // ---- stage 1 ----
  attn_rows_kernel<<<gAttn, blk, 0, stream>>>(x, 10, Wq1, Wk1, Wv1, Wo1, hA, B);

  hipMemsetAsync(gsm, 0, 512 * sizeof(float), stream);
  gemm_kernel<32, 32, 128, 0, 2, false, true, _Float16>
      <<<dim3(gx, 8), blk, 0, stream>>>(hA, nullptr, nullptr, wtA1, bpA1, za, gsm, gsq);
  bn_coef_kernel<<<1, 256, 0, stream>>>(gsm, gsq, g1a, be1a, ca, cc, 128, invB);

  hipMemsetAsync(gsm, 0, 512 * sizeof(float), stream);
  gemm_kernel<128, 128, 256, 0, 2, true, true, _Float16>
      <<<dim3(gx, 16), blk, 0, stream>>>(za, ca, cc, wtB1, bpB1, zb, gsm, gsq);
  bn_coef_kernel<<<1, 256, 0, stream>>>(gsm, gsq, g1b, be1b, ca, cc, 256, invB);

  gemm_kernel<256, 256, 32, 0, 2, true, false, float>
      <<<dim3(gx, 1), blk, 0, stream>>>(zb, ca, cc, wtC1, bpC1, hz, nullptr, nullptr);

  // ---- stage 2 ----
  attn_rows_kernel<<<gAttn, blk, 0, stream>>>(hz, 32, Wq2, Wk2, Wv2, Wo2, hA, B);

  hipMemsetAsync(gsm, 0, 512 * sizeof(float), stream);
  gemm_kernel<32, 32, 128, 0, 2, false, true, _Float16>
      <<<dim3(gx, 8), blk, 0, stream>>>(hA, nullptr, nullptr, wtA2, bpA2, za, gsm, gsq);
  bn_coef_kernel<<<1, 256, 0, stream>>>(gsm, gsq, g2a, be2a, ca, cc, 128, invB);

  hipMemsetAsync(gsm, 0, 512 * sizeof(float), stream);
  gemm_kernel<128, 128, 256, 0, 2, true, true, _Float16>
      <<<dim3(gx, 16), blk, 0, stream>>>(za, ca, cc, wtB2, bpB2, zb, gsm, gsq);
  bn_coef_kernel<<<1, 256, 0, stream>>>(gsm, gsq, g2b, be2b, ca, cc, 256, invB);

  gemm_kernel<256, 256, 2, 2, 2, true, false, float>
      <<<dim3(gx, 1), blk, 0, stream>>>(zb, ca, cc, wtC2, bpC2, (float*)d_out, nullptr, nullptr);
}
